// SelfAttentionHead_33956011442433
// MI455X (gfx1250) — matrix operand order repaired and checked by reading, not compiled, not run on hardware
//
#include <hip/hip_runtime.h>

// ---------------------------------------------------------------------------
// MI455X (gfx1250) fused single-head causal attention
//   B=8, S=2048, E=1024, H=64
//   k0: weight transpose+convert  (fp32 [E][H] -> f16 [3][H][E], 0.75 MB once)
//   k1: QKV projection; x read once from HBM via async global->LDS b128,
//       double-buffered (prefetch tile k+1 during WMMA on tile k),
//       f16 WMMA w/ f32 accum, V written pre-transposed [b][h][s]
//   k2: causal flash attention; K/V tiles double-buffered async LDS copies,
//       heavy (late-q) blocks dispatched first
// ---------------------------------------------------------------------------

typedef __attribute__((ext_vector_type(16))) _Float16 v16h;
typedef __attribute__((ext_vector_type(8)))  float    v8f;
typedef int v4i_vs __attribute__((vector_size(16)));   // matches builtin param

#define WMMA_F16(a, b, c) \
  __builtin_amdgcn_wmma_f32_16x16x32_f16(false, (a), false, (b), (short)0, (c), false, false)

static constexpr int B_  = 8;
static constexpr int S_  = 2048;
static constexpr int E_  = 1024;
static constexpr int H_  = 64;
static constexpr long N_ = (long)B_ * S_;   // 16384 rows

// --------------------- CDNA5 async global->LDS copy path -------------------
#if defined(__has_builtin)
#if __has_builtin(__builtin_amdgcn_global_load_async_to_lds_b128) && \
    __has_builtin(__builtin_amdgcn_s_wait_asynccnt)
#define USE_ASYNC_LDS 1
#endif
#endif
#ifndef USE_ASYNC_LDS
#define USE_ASYNC_LDS 0
#endif

#if USE_ASYNC_LDS
// builtin signature (from clang diagnostic): (v4i AS1*, v4i AS3*, imm, imm)
#define ASYNC_COPY16(gsrc, ldst)                                                         \
  __builtin_amdgcn_global_load_async_to_lds_b128(                                        \
      (__attribute__((address_space(1))) v4i_vs*)(unsigned long long)(const void*)(gsrc),\
      (__attribute__((address_space(3))) v4i_vs*)(unsigned long long)(void*)(ldst),      \
      0, 0)
#define WAIT_ASYNC(n) __builtin_amdgcn_s_wait_asynccnt(n)
#else
#define WAIT_ASYNC(n)
#endif

// ---------------------------------------------------------------------------
// Fragment gathers matching CDNA5 WMMA VGPR layouts (ISA 7.12.2), f16 source.
// A (16x32): lane (r=L&15,hi=L>>4) holds row r, elems 0..7 = K hi*8..+7,
//            elems 8..15 = K 16+hi*8..+7  -> two b128 loads.
// B (32x16) from TRANSPOSED tile Bt[n][k]: lane n holds K hi*8..+7, 16+hi*8..+7.
// ---------------------------------------------------------------------------
__device__ __forceinline__ v16h load_frag_a(const _Float16* base, int ld,
                                            int row0, int kbase, int lane) {
  const int r = lane & 15, hi = lane >> 4;
  const _Float16* p = base + (long)(row0 + r) * ld + kbase + hi * 8;
  v16h a;
  *((uint4*)&a)     = *(const uint4*)p;
  *((uint4*)&a + 1) = *(const uint4*)(p + 16);
  return a;
}

__device__ __forceinline__ v16h load_frag_b(const _Float16* bt, int ld,
                                            int n0, int kbase, int lane) {
  const int n = lane & 15, hi = lane >> 4;
  const _Float16* p = bt + (long)(n0 + n) * ld + kbase + hi * 8;
  v16h b;
  *((uint4*)&b)     = *(const uint4*)p;
  *((uint4*)&b + 1) = *(const uint4*)(p + 16);
  return b;
}

// A fragment from an fp32 LDS tile (convert at fragment build; the compiler
// fuses these into v_cvt_pk_f16_f32 pairs, co-executing with WMMA).
__device__ __forceinline__ v16h load_frag_a_f32(const float* base, int ld,
                                                int row0, int kbase, int lane) {
  const int r = lane & 15, hi = lane >> 4;
  const float* p = base + (long)(row0 + r) * ld + kbase + hi * 8;
  float4 f0 = *(const float4*)p;
  float4 f1 = *(const float4*)(p + 4);
  float4 f2 = *(const float4*)(p + 16);
  float4 f3 = *(const float4*)(p + 20);
  v16h a;
  a[0]  = (_Float16)f0.x; a[1]  = (_Float16)f0.y; a[2]  = (_Float16)f0.z; a[3]  = (_Float16)f0.w;
  a[4]  = (_Float16)f1.x; a[5]  = (_Float16)f1.y; a[6]  = (_Float16)f1.z; a[7]  = (_Float16)f1.w;
  a[8]  = (_Float16)f2.x; a[9]  = (_Float16)f2.y; a[10] = (_Float16)f2.z; a[11] = (_Float16)f2.w;
  a[12] = (_Float16)f3.x; a[13] = (_Float16)f3.y; a[14] = (_Float16)f3.z; a[15] = (_Float16)f3.w;
  return a;
}

// ---------------------------------------------------------------------------
// Kernel 0: weight transpose + convert: Wt[m][n][e] = (f16) W_m[e][n]
// ---------------------------------------------------------------------------
__global__ __launch_bounds__(256) void wtrans_kernel(
    const float* __restrict__ Wq, const float* __restrict__ Wk,
    const float* __restrict__ Wv, _Float16* __restrict__ Wtg) {
  const float* Ws[3] = {Wq, Wk, Wv};
  const float* W = Ws[blockIdx.y];
  const long mbase = (long)blockIdx.y * H_ * E_;
  const int  base  = blockIdx.x * 1024;
  #pragma unroll
  for (int i = 0; i < 4; ++i) {
    int o = base + threadIdx.x + i * 256;   // o = n*1024 + e
    int n = o >> 10, e = o & 1023;
    Wtg[mbase + o] = (_Float16)W[(long)e * H_ + n];
  }
}

// ---------------------------------------------------------------------------
// Kernel 1: QKV projection.  Block = 128 threads (4 waves), 64 rows of x.
// Double-buffered async tile pipeline: 20 b128 copies / thread / tile.
// ---------------------------------------------------------------------------
#define PJ_TM  64
#define PJ_TK  64
#define PJ_LDX 68   // fp32 stride: 16B-aligned chunks, rows 4 banks apart
#define PJ_LDW 72   // f16 stride
#define PJ_NK  (E_ / PJ_TK)       // 16 k-steps

__device__ __forceinline__ void pj_issue_tile(
    const float* __restrict__ x, const _Float16* __restrict__ Wtg,
    long r0, int kk, int tid, float* Xbuf, _Float16* Wbuf) {
#if USE_ASYNC_LDS
  #pragma unroll
  for (int i = 0; i < 8; ++i) {             // x: 64 rows x 16 chunks(4 f32)
    int c = tid + i * 128, row = c >> 4, c4 = (c & 15) * 4;
    ASYNC_COPY16(x + (r0 + row) * E_ + kk + c4, Xbuf + row * PJ_LDX + c4);
  }
  #pragma unroll
  for (int i = 0; i < 12; ++i) {            // W: 3 x 64 rows x 8 chunks(8 f16)
    int c = tid + i * 128, m = c >> 9, cm = c & 511;
    int n = cm >> 3, k8 = (cm & 7) * 8;
    ASYNC_COPY16(Wtg + (long)m * H_ * E_ + (long)n * E_ + kk + k8,
                 Wbuf + m * (PJ_TK * PJ_LDW) + n * PJ_LDW + k8);
  }
#else
  uint4 xr[8], wr[12];                      // two-phase: batch loads, then store
  #pragma unroll
  for (int i = 0; i < 8; ++i) {
    int c = tid + i * 128, row = c >> 4, c4 = (c & 15) * 4;
    xr[i] = *(const uint4*)(x + (r0 + row) * E_ + kk + c4);
  }
  #pragma unroll
  for (int i = 0; i < 12; ++i) {
    int c = tid + i * 128, m = c >> 9, cm = c & 511;
    int n = cm >> 3, k8 = (cm & 7) * 8;
    wr[i] = *(const uint4*)(Wtg + (long)m * H_ * E_ + (long)n * E_ + kk + k8);
  }
  #pragma unroll
  for (int i = 0; i < 8; ++i) {
    int c = tid + i * 128, row = c >> 4, c4 = (c & 15) * 4;
    *(uint4*)(Xbuf + row * PJ_LDX + c4) = xr[i];
  }
  #pragma unroll
  for (int i = 0; i < 12; ++i) {
    int c = tid + i * 128, m = c >> 9, cm = c & 511;
    int n = cm >> 3, k8 = (cm & 7) * 8;
    *(uint4*)(Wbuf + m * (PJ_TK * PJ_LDW) + n * PJ_LDW + k8) = wr[i];
  }
#endif
}

__global__ __launch_bounds__(128) void qkv_proj_kernel(
    const float* __restrict__ x, const _Float16* __restrict__ Wtg,
    _Float16* __restrict__ Qh, _Float16* __restrict__ Kh,
    _Float16* __restrict__ Vtg) {
  __shared__ float    Xs[2][PJ_TM * PJ_LDX];      // 2 x 17.4 KB
  __shared__ _Float16 Wt[2][3 * PJ_TK * PJ_LDW];  // 2 x 27.6 KB

  const int tid  = threadIdx.x;
  const int lane = tid & 31;
  const int w    = tid >> 5;                  // wave: rows w*16..w*16+15
  const long r0  = (long)blockIdx.x * PJ_TM;

  v8f acc[3][4] = {};

  pj_issue_tile(x, Wtg, r0, 0, tid, Xs[0], Wt[0]);       // prefetch tile 0

  for (int it = 0; it < PJ_NK; ++it) {
    const float*    Xcur = Xs[it & 1];
    const _Float16* Wcur = Wt[it & 1];
    if (it + 1 < PJ_NK) {                   // prefetch next tile, overlap WMMA
      pj_issue_tile(x, Wtg, r0, (it + 1) * PJ_TK, tid,
                    Xs[(it + 1) & 1], Wt[(it + 1) & 1]);
      WAIT_ASYNC(20);                       // oldest 20 (tile `it`) landed
    } else {
      WAIT_ASYNC(0);
    }
    __syncthreads();                        // tile `it` collectively visible

    v16h a0 = load_frag_a_f32(Xcur, PJ_LDX, w * 16, 0,  lane);
    v16h a1 = load_frag_a_f32(Xcur, PJ_LDX, w * 16, 32, lane);
    #pragma unroll
    for (int m = 0; m < 3; ++m) {
      #pragma unroll
      for (int nt = 0; nt < 4; ++nt) {
        v16h b0 = load_frag_b(Wcur + m * (PJ_TK * PJ_LDW), PJ_LDW, nt * 16, 0,  lane);
        v16h b1 = load_frag_b(Wcur + m * (PJ_TK * PJ_LDW), PJ_LDW, nt * 16, 32, lane);
        acc[m][nt] = WMMA_F16(a0, b0, acc[m][nt]);
        acc[m][nt] = WMMA_F16(a1, b1, acc[m][nt]);
      }
    }
    __syncthreads();                        // buffer free for prefetch of it+2
  }

  // Epilogue.  D layout: lane holds col n = lane&15, rows hi*8+v.
  const int n = lane & 15, hi = lane >> 4;
  _Float16* QK[2] = {Qh, Kh};
  #pragma unroll
  for (int m = 0; m < 2; ++m)
    #pragma unroll
    for (int nt = 0; nt < 4; ++nt)
      #pragma unroll
      for (int v = 0; v < 8; ++v) {
        long row = r0 + w * 16 + hi * 8 + v;
        QK[m][row * H_ + nt * 16 + n] = (_Float16)acc[m][nt][v];
      }
  // V transposed [b][h][s]: lane's 8 rows are consecutive s -> one b128 store.
  const long rowg = r0 + w * 16 + hi * 8;       // 64-row tile never crosses batch
  const int  bb   = (int)(rowg >> 11);
  const long s0   = rowg & (S_ - 1);
  #pragma unroll
  for (int nt = 0; nt < 4; ++nt) {
    int col = nt * 16 + n;
    union { uint4 u; _Float16 h[8]; } t;
    #pragma unroll
    for (int v = 0; v < 8; ++v) t.h[v] = (_Float16)acc[2][nt][v];
    *(uint4*)(Vtg + ((long)bb * H_ + col) * S_ + s0) = t.u;
  }
}

// ---------------------------------------------------------------------------
// Kernel 2: causal flash attention.  Block = (b, 64 q-rows), 4 waves.
// Double-buffered async K/V tiles: 8 b128 copies / thread / tile.
// Heavy blocks (large q0, most key tiles) are dispatched FIRST.
// ---------------------------------------------------------------------------
#define AT_MQ  64
#define AT_KT  64
#define AT_LDK 72
#define AT_LDV 72
#define AT_LDP 72
#define AT_NQ  (S_ / AT_MQ)   // 32 q-tiles per batch

__device__ __forceinline__ void at_issue_tile(
    const _Float16* __restrict__ Kh, const _Float16* __restrict__ Vtg,
    long bbase, int k0, int tid, _Float16* Kbuf, _Float16* Vbuf) {
#if USE_ASYNC_LDS
  #pragma unroll
  for (int i = 0; i < 4; ++i) {             // K: 64 keys x 8 chunks(8 f16)
    int c = tid + i * 128, key = c >> 3, h8 = (c & 7) * 8;
    ASYNC_COPY16(Kh + bbase + (long)(k0 + key) * H_ + h8,
                 Kbuf + key * AT_LDK + h8);
  }
  #pragma unroll
  for (int i = 0; i < 4; ++i) {             // V: 64 h-rows x 8 chunks(8 f16)
    int c = tid + i * 128, h = c >> 3, s8 = (c & 7) * 8;
    ASYNC_COPY16(Vtg + bbase + (long)h * S_ + k0 + s8,
                 Vbuf + h * AT_LDV + s8);
  }
#else
  uint4 kr[4], vr[4];
  #pragma unroll
  for (int i = 0; i < 4; ++i) {
    int c = tid + i * 128, key = c >> 3, h8 = (c & 7) * 8;
    kr[i] = *(const uint4*)(Kh + bbase + (long)(k0 + key) * H_ + h8);
  }
  #pragma unroll
  for (int i = 0; i < 4; ++i) {
    int c = tid + i * 128, h = c >> 3, s8 = (c & 7) * 8;
    vr[i] = *(const uint4*)(Vtg + bbase + (long)h * S_ + k0 + s8);
  }
  #pragma unroll
  for (int i = 0; i < 4; ++i) {
    int c = tid + i * 128, key = c >> 3, h8 = (c & 7) * 8;
    *(uint4*)(Kbuf + key * AT_LDK + h8) = kr[i];
  }
  #pragma unroll
  for (int i = 0; i < 4; ++i) {
    int c = tid + i * 128, h = c >> 3, s8 = (c & 7) * 8;
    *(uint4*)(Vbuf + h * AT_LDV + s8) = vr[i];
  }
#endif
}

__global__ __launch_bounds__(128) void attn_kernel(
    const _Float16* __restrict__ Qh, const _Float16* __restrict__ Kh,
    const _Float16* __restrict__ Vtg, float* __restrict__ out) {
  __shared__ _Float16 Kt[2][AT_KT * AT_LDK];  // [key][h]
  __shared__ _Float16 Vt[2][H_ * AT_LDV];     // [h][key]
  __shared__ _Float16 Pl[4][16 * AT_LDP];     // per-wave P scratch

  const int tid  = threadIdx.x;
  const int lane = tid & 31;
  const int w    = tid >> 5;
  const int n    = lane & 15, hi = lane >> 4;

  const int  b     = blockIdx.y;
  const int  qt    = (AT_NQ - 1) - blockIdx.x;   // heavy tiles first
  const long q0    = (long)qt * AT_MQ;
  const long bbase = (long)b * S_ * H_;       // same stride for Qh/Kh/Vtg/out
  const int  qw0   = (int)q0 + w * 16;

  const _Float16* qptr = Qh + bbase + (long)qw0 * H_;
  v16h qa0 = load_frag_a(qptr, H_, 0, 0,  lane);
  v16h qa1 = load_frag_a(qptr, H_, 0, 32, lane);
  const _Float16 sc = (_Float16)0.125f;       // 1/sqrt(64), exact in f16
  qa0 *= sc;
  qa1 *= sc;

  v8f   o[4] = {};
  float mrun[8], lrun[8];
  #pragma unroll
  for (int v = 0; v < 8; ++v) { mrun[v] = -1e30f; lrun[v] = 0.0f; }

  const int jtiles = qt + 1;                  // causal: keys <= q0+63

  at_issue_tile(Kh, Vtg, bbase, 0, tid, Kt[0], Vt[0]);   // prefetch tile 0

  for (int j = 0; j < jtiles; ++j) {
    const _Float16* Kcur = Kt[j & 1];
    const _Float16* Vcur = Vt[j & 1];
    if (j + 1 < jtiles) {                   // prefetch next K/V tile
      at_issue_tile(Kh, Vtg, bbase, (j + 1) * AT_KT, tid,
                    Kt[(j + 1) & 1], Vt[(j + 1) & 1]);
      WAIT_ASYNC(8);                        // oldest 8 (tile j) landed
    } else {
      WAIT_ASYNC(0);
    }
    __syncthreads();                        // tile j collectively visible
    const int k0 = j * AT_KT;

    // --- scores S = Q.K^T: 4 subtiles of 16 keys, 2 WMMA each (K=64) ---
    float s[4][8];
    #pragma unroll
    for (int t = 0; t < 4; ++t) {
      v8f cacc = {};
      v16h bk0 = load_frag_b(Kcur, AT_LDK, t * 16, 0,  lane);
      v16h bk1 = load_frag_b(Kcur, AT_LDK, t * 16, 32, lane);
      cacc = WMMA_F16(qa0, bk0, cacc);
      cacc = WMMA_F16(qa1, bk1, cacc);
      const int key = k0 + t * 16 + n;
      #pragma unroll
      for (int v = 0; v < 8; ++v) {
        const int q = qw0 + hi * 8 + v;
        s[t][v] = (key > q) ? -1e30f : cacc[v];   // causal mask
      }
    }

    // --- online softmax: row hi*8+v lives on one 16-lane half ---
    float p[4][8];
    #pragma unroll
    for (int v = 0; v < 8; ++v) {
      float mx = fmaxf(fmaxf(s[0][v], s[1][v]), fmaxf(s[2][v], s[3][v]));
      #pragma unroll
      for (int d = 1; d < 16; d <<= 1) mx = fmaxf(mx, __shfl_xor(mx, d, 32));
      const float mnew  = fmaxf(mrun[v], mx);
      const float alpha = __expf(mrun[v] - mnew);
      mrun[v] = mnew;
      float rowsum = 0.0f;
      #pragma unroll
      for (int t = 0; t < 4; ++t) {
        float e = __expf(s[t][v] - mnew);
        p[t][v] = e;
        rowsum += e;
      }
      #pragma unroll
      for (int d = 1; d < 16; d <<= 1) rowsum += __shfl_xor(rowsum, d, 32);
      lrun[v] = lrun[v] * alpha + rowsum;
      #pragma unroll
      for (int nt = 0; nt < 4; ++nt) o[nt][v] *= alpha;
    }

    // --- reshape P: D layout -> A layout via per-wave LDS scratch ---
    _Float16* pw = Pl[w];
    #pragma unroll
    for (int t = 0; t < 4; ++t)
      #pragma unroll
      for (int v = 0; v < 8; ++v)
        pw[(hi * 8 + v) * AT_LDP + t * 16 + n] = (_Float16)p[t][v];
    __syncthreads();

    v16h pa0 = load_frag_a(pw, AT_LDP, 0, 0,  lane);
    v16h pa1 = load_frag_a(pw, AT_LDP, 0, 32, lane);
    #pragma unroll
    for (int nt = 0; nt < 4; ++nt) {
      v16h bv0 = load_frag_b(Vcur, AT_LDV, nt * 16, 0,  lane);
      v16h bv1 = load_frag_b(Vcur, AT_LDV, nt * 16, 32, lane);
      o[nt] = WMMA_F16(pa0, bv0, o[nt]);
      o[nt] = WMMA_F16(pa1, bv1, o[nt]);
    }
    __syncthreads();                        // K/V buffer free for prefetch
  }

  // --- epilogue: out = O / l, fp32, non-temporal (write-once stream) ---
  #pragma unroll
  for (int v = 0; v < 8; ++v) {
    const float inv = 1.0f / lrun[v];
    const long  q   = q0 + w * 16 + hi * 8 + v;
    float* op = out + bbase + q * H_;
    #pragma unroll
    for (int nt = 0; nt < 4; ++nt)
      __builtin_nontemporal_store(o[nt][v] * inv, op + nt * 16 + n);
  }
}

// ---------------------------------------------------------------------------
// Host launch
// ---------------------------------------------------------------------------
extern "C" void kernel_launch(void* const* d_in, const int* in_sizes, int n_in,
                              void* d_out, int out_size, void* d_ws, size_t ws_size,
                              hipStream_t stream) {
  const float* x  = (const float*)d_in[0];   // (B,S,E)
  const float* Wk = (const float*)d_in[1];   // (E,H)
  const float* Wq = (const float*)d_in[2];   // (E,H)
  const float* Wv = (const float*)d_in[3];   // (E,H)
  float* out = (float*)d_out;                // (B,S,H) fp32

  _Float16* Qh  = (_Float16*)d_ws;           // [N][H]      2 MB
  _Float16* Kh  = Qh + N_ * H_;              // [N][H]      2 MB
  _Float16* Vtg = Kh + N_ * H_;              // [B][H][S]   2 MB (transposed)
  _Float16* Wtg = Vtg + N_ * H_;             // [3][H][E]   384 KB

  wtrans_kernel<<<dim3(64, 3), 256, 0, stream>>>(Wq, Wk, Wv, Wtg);
  qkv_proj_kernel<<<dim3((int)(N_ / PJ_TM)), 128, 0, stream>>>(x, Wtg, Qh, Kh, Vtg);
  attn_kernel<<<dim3(AT_NQ, B_), 128, 0, stream>>>(Qh, Kh, Vtg, out);
}
